// TransformerBlock3D_24550033064609
// MI455X (gfx1250) — matrix operand order repaired and checked by reading, not compiled, not run on hardware
//
#include <hip/hip_runtime.h>
#include <hip/hip_bf16.h>
#include <cstdint>

// ---------------------------------------------------------------------------
// TransformerBlock3D for MI455X (gfx1250, wave32, WMMA)
// B=8, C=128, S=32 -> NTOK=32768 tokens/batch, HEADS=8, dim_head=16, MLP=512
// ---------------------------------------------------------------------------

#define NTOK 32768
#define CDIM 128
#define BDIM 8
#define MLPD 512
#define LN_BLOCKS 64

typedef __attribute__((ext_vector_type(16))) __bf16 v16bf;
typedef __attribute__((ext_vector_type(8)))  float  v8f;
typedef __attribute__((ext_vector_type(2)))  float  v2f;

union FragAB { v16bf v; uint4 q[2]; };

__device__ __forceinline__ unsigned short f32_to_bf16(float f) {
  unsigned int u = __float_as_uint(f);
  unsigned int r = u + 0x7FFFu + ((u >> 16) & 1u);   // round-to-nearest-even
  return (unsigned short)(r >> 16);
}

// CDNA5 async global->LDS copy (ASYNCcnt-tracked), 16B per lane.
__device__ __forceinline__ void async_g2l_b128(void* ldsDst, const void* gsrc) {
  unsigned loff = (unsigned)(uintptr_t)ldsDst;   // low 32 bits = LDS byte offset
  asm volatile("global_load_async_to_lds_b128 %0, %1, off"
               :: "v"(loff), "v"(gsrc) : "memory");
}
__device__ __forceinline__ void wait_async0() {
  asm volatile("s_wait_asynccnt 0x0" ::: "memory");
}

// --------------------------- weight f32 -> bf16 ----------------------------
__global__ __launch_bounds__(256)
void cvt_f32_bf16_kernel(const float* __restrict__ src,
                         unsigned short* __restrict__ dst, int n) {
  int i = blockIdx.x * 256 + threadIdx.x;
  if (i < n) dst[i] = f32_to_bf16(src[i]);
}

// ---------------- channel-major f32 -> token-major bf16 (LDS tiles) --------
__global__ __launch_bounds__(256)
void transpose_cm_to_tm_bf16(const float* __restrict__ src, long long srcBatchStride,
                             int C, unsigned short* __restrict__ dst) {
  __shared__ float tile[32][33];
  const int b     = blockIdx.z;
  const int nBase = blockIdx.x * 32;
  const int cBase = blockIdx.y * 32;
  const int tx = threadIdx.x & 31;
  const int ty = threadIdx.x >> 5;          // 0..7
  const float* s = src + (size_t)b * (size_t)srcBatchStride;
#pragma unroll
  for (int i = 0; i < 4; ++i) {
    int c = cBase + ty + i * 8;
    tile[ty + i * 8][tx] = s[(size_t)c * NTOK + nBase + tx];
  }
  __syncthreads();
#pragma unroll
  for (int i = 0; i < 4; ++i) {
    int n = nBase + ty + i * 8;
    dst[((size_t)b * NTOK + n) * C + cBase + tx] = f32_to_bf16(tile[tx][ty + i * 8]);
  }
}

// ------------------------- main bf16 WMMA token GEMM -----------------------
// Y[b][o][tok] = sum_k W[b?][o][k] * Xtm[b][tok][k]  (+bias)(gelu?)(+resid)
// Block: 256 thr = 8 waves; wave tile = 64(O) x 32(tok); block = 64(O) x 256(tok)
// W tile (64 x K bf16) staged to LDS via async global->LDS; each LDS A-fragment
// feeds two WMMAs (two token sub-tiles) so per-K-step a wave issues 8 WMMAs for
// 4 global B-loads + 8 LDS reads. B stream double-buffered in registers.
// Grid: (NTOK/256, O/64, B)
template <int K>
__global__ __launch_bounds__(256)
void gemm_bf16_wmma(const unsigned short* __restrict__ W, long long wBatchStride,
                    const unsigned short* __restrict__ Xtm, int O,
                    const float* __restrict__ bias,
                    const float* __restrict__ resid,   // [b][O][NTOK] f32 or null
                    float* __restrict__ outCM,         // [b][O][NTOK] f32 or null
                    unsigned short* __restrict__ outTM,// [b][NTOK][O] bf16 or null
                    int doGelu) {
  constexpr int KP = K + 8;                 // +16B row pad: bank spread, 16B align
  constexpr int LW = (64 * KP > 256 * 64) ? 64 * KP : 256 * 64;
  __shared__ unsigned short lw[LW];

  const int tid  = threadIdx.x;
  const int lane = tid & 31;
  const int wave = tid >> 5;
  const int b    = blockIdx.z;
  const int tokBase0 = blockIdx.x * 256;
  const int tokBase  = tokBase0 + wave * 32;
  const int oBase    = blockIdx.y * 64;
  const int hi = lane >> 4;                 // 0/1
  const int lo = lane & 15;
  const int khalf = hi * 16;

  // ---- stage W tile into LDS (async, ASYNCcnt) ----
  {
    const unsigned short* wbase =
        W + (size_t)b * (size_t)wBatchStride + (size_t)oBase * K;
    constexpr int UNITS = 64 * K / 8;       // 16B units
#pragma unroll
    for (int i = tid; i < UNITS; i += 256) {
      int row = i / (K / 8), c8 = i % (K / 8);
      async_g2l_b128(&lw[row * KP + c8 * 8], wbase + (size_t)row * K + c8 * 8);
    }
    wait_async0();
  }
  __syncthreads();

  // ---- K loop: LDS A-fragments, double-buffered global B-fragments ----
  const unsigned short* xrow0 =
      Xtm + ((size_t)b * NTOK + (tokBase + lo)) * K + hi * 16;
  const unsigned short* xrow1 = xrow0 + (size_t)16 * K;

  v8f acc[4][2] = {};
  FragAB bcur[2], bnext[2];
  bcur[0].q[0] = *(const uint4*)(xrow0);
  bcur[0].q[1] = *(const uint4*)(xrow0 + 8);
  bcur[1].q[0] = *(const uint4*)(xrow1);
  bcur[1].q[1] = *(const uint4*)(xrow1 + 8);

#pragma unroll 2
  for (int kk = 0; kk < K; kk += 32) {
    const bool more = (kk + 32) < K;
    if (more) {
      bnext[0].q[0] = *(const uint4*)(xrow0 + kk + 32);
      bnext[0].q[1] = *(const uint4*)(xrow0 + kk + 40);
      bnext[1].q[0] = *(const uint4*)(xrow1 + kk + 32);
      bnext[1].q[1] = *(const uint4*)(xrow1 + kk + 40);
      __builtin_prefetch(xrow0 + kk + 64, 0, 1);
      __builtin_prefetch(xrow1 + kk + 64, 0, 1);
    }
#pragma unroll
    for (int s = 0; s < 4; ++s) {
      // A (16x32 bf16): row m = lo, contiguous chunk [khalf,+16)
      const int lofs = (s * 16 + lo) * KP + kk + khalf;
      FragAB a;
      a.q[0] = *(const uint4*)(&lw[lofs]);
      a.q[1] = *(const uint4*)(&lw[lofs + 8]);
      acc[s][0] = __builtin_amdgcn_wmma_f32_16x16x32_bf16(
          false, a.v, false, bcur[0].v, (short)0, acc[s][0], false, false);
      acc[s][1] = __builtin_amdgcn_wmma_f32_16x16x32_bf16(
          false, a.v, false, bcur[1].v, (short)0, acc[s][1], false, false);
    }
    if (more) { bcur[0] = bnext[0]; bcur[1] = bnext[1]; }
  }

  if (outTM) {
    // ---- transposed bf16 epilogue, coalesced through LDS [256 tok][64 o] ----
    __syncthreads();                        // done reading lw; reuse as staging
#pragma unroll
    for (int t = 0; t < 2; ++t) {
#pragma unroll
      for (int s = 0; s < 4; ++s) {
#pragma unroll
        for (int r = 0; r < 8; ++r) {
          int ol = s * 16 + hi * 8 + r;     // D layout: M = r + hi*8
          float v = acc[s][t][r];
          if (bias)   v += bias[oBase + ol];
          if (doGelu) v = 0.5f * v * (1.0f + erff(v * 0.70710678118654752f));
          lw[(wave * 32 + t * 16 + lo) * 64 + ol] = f32_to_bf16(v);
        }
      }
    }
    __syncthreads();
    for (int u = tid; u < 256 * 64 / 8; u += 256) {
      int row = u >> 3, c8 = u & 7;
      *(uint4*)(outTM + ((size_t)b * NTOK + tokBase0 + row) * O + oBase + c8 * 8) =
          *(const uint4*)(&lw[row * 64 + c8 * 8]);
    }
  } else {
    // ---- channel-major f32 epilogue (64B-coalesced per half-wave) ----
#pragma unroll
    for (int t = 0; t < 2; ++t) {
      const int tok = tokBase + t * 16 + lo;
#pragma unroll
      for (int s = 0; s < 4; ++s) {
#pragma unroll
        for (int r = 0; r < 8; ++r) {
          int o = oBase + s * 16 + hi * 8 + r;
          float v = acc[s][t][r];
          if (bias)   v += bias[o];
          if (doGelu) v = 0.5f * v * (1.0f + erff(v * 0.70710678118654752f));
          if (resid)  v += resid[((size_t)b * O + o) * NTOK + tok];
          outCM[((size_t)b * O + o) * NTOK + tok] = v;
        }
      }
    }
  }
}

// -------- attention scores (fp32 WMMA 16x16x4, K = 32768 tokens) + softmax --
// One block per (b, head). qkv is [b][384][NTOK] f32; attn out [b][h][16][16].
// Uniform trip count keeps EXEC structurally all-ones around WMMA (scalar
// loop branch); two interleaved accumulators break the WMMA RAW chain.
__global__ __launch_bounds__(256)
void attn_scores_softmax(const float* __restrict__ qkv, float* __restrict__ attn) {
  __shared__ float red[8 * 256];
  __shared__ float smat[256];
  const int bh = blockIdx.x;
  const int b = bh >> 3, h = bh & 7;
  const int lane = threadIdx.x & 31, wave = threadIdx.x >> 5;
  const int hi = lane >> 4, lo = lane & 15;

  const float* q = qkv + ((size_t)b * 384 + h * 16) * NTOK + (size_t)lo * NTOK
                       + wave * 8 + hi * 2;
  const float* k = qkv + ((size_t)b * 384 + 128 + h * 16) * NTOK + (size_t)lo * NTOK
                       + wave * 8 + hi * 2;

  v8f acc0 = {}, acc1 = {};
#pragma unroll 2
  for (int it = 0; it < NTOK / 64; ++it) {        // uniform scalar trip count
    const int tt = it * 64;
    v2f a0 = *(const v2f*)(q + tt);
    v2f b0 = *(const v2f*)(k + tt);
    v2f a1 = *(const v2f*)(q + tt + 4);
    v2f b1 = *(const v2f*)(k + tt + 4);
    acc0 = __builtin_amdgcn_wmma_f32_16x16x4_f32(
        false, a0, false, b0, (short)0, acc0, false, false);
    acc1 = __builtin_amdgcn_wmma_f32_16x16x4_f32(
        false, a1, false, b1, (short)0, acc1, false, false);
  }
#pragma unroll
  for (int r = 0; r < 8; ++r) red[wave * 256 + r * 32 + lane] = acc0[r] + acc1[r];
  __syncthreads();

  // combine 8 wave partials; thread t -> (i = t>>4, j = t&15)
  const int i = threadIdx.x >> 4, j = threadIdx.x & 15;
  float s = 0.f;
#pragma unroll
  for (int w = 0; w < 8; ++w)
    s += red[w * 256 + (i & 7) * 32 + (i >> 3) * 16 + j];
  smat[i * 16 + j] = s * 0.25f;                   // SCALE = 16^-0.5
  __syncthreads();

  if (threadIdx.x < 16) {
    int row = threadIdx.x;
    float mx = -1e30f;
    for (int c = 0; c < 16; ++c) mx = fmaxf(mx, smat[row * 16 + c]);
    float e[16]; float sum = 0.f;
    for (int c = 0; c < 16; ++c) { e[c] = expf(smat[row * 16 + c] - mx); sum += e[c]; }
    float inv = 1.0f / sum;
    float* a = attn + (size_t)(bh)*256 + row * 16;
    for (int c = 0; c < 16; ++c) a[c] = e[c] * inv;
  }
}

// ------- fuse: M_b[o][h*16+j] = sum_i proj_w[o][h*16+i] * attn[b][h][i][j] ---
__global__ __launch_bounds__(256)
void fuse_proj_attn(const float* __restrict__ proj_w, const float* __restrict__ attn,
                    unsigned short* __restrict__ Mb) {
  const int b = blockIdx.x;
  for (int idx = threadIdx.x; idx < CDIM * CDIM; idx += 256) {
    int o = idx >> 7, cc = idx & 127;
    int h = cc >> 4, j = cc & 15;
    const float* a = attn + (size_t)(b * 8 + h) * 256 + j;
    const float* w = proj_w + o * CDIM + h * 16;
    float s = 0.f;
#pragma unroll
    for (int i = 0; i < 16; ++i) s += w[i] * a[i * 16];
    Mb[(size_t)b * (CDIM * CDIM) + idx] = f32_to_bf16(s);
  }
}

// ------------------------- full-tensor LayerNorm ---------------------------
__global__ __launch_bounds__(256)
void ln_reduce(const float* __restrict__ x, float2* __restrict__ partials) {
  const int b = blockIdx.y, blk = blockIdx.x;
  const size_t per = (size_t)CDIM * NTOK;
  const size_t chunk = per / LN_BLOCKS;           // 65536 floats
  const float4* p4 = (const float4*)(x + (size_t)b * per + (size_t)blk * chunk);
  float s = 0.f, ss = 0.f;
  for (int i = threadIdx.x; i < (int)(chunk / 4); i += 256) {
    float4 v = p4[i];
    s  += v.x + v.y + v.z + v.w;
    ss += v.x * v.x + v.y * v.y + v.z * v.z + v.w * v.w;
  }
  __shared__ float rs[256], rss[256];
  rs[threadIdx.x] = s; rss[threadIdx.x] = ss; __syncthreads();
  for (int st = 128; st > 0; st >>= 1) {
    if ((int)threadIdx.x < st) { rs[threadIdx.x] += rs[threadIdx.x + st];
                                 rss[threadIdx.x] += rss[threadIdx.x + st]; }
    __syncthreads();
  }
  if (threadIdx.x == 0) {
    float2 o; o.x = rs[0]; o.y = rss[0];
    partials[b * LN_BLOCKS + blk] = o;
  }
}

__global__ __launch_bounds__(64)
void ln_finalize(const float2* __restrict__ partials, float2* __restrict__ stats) {
  __shared__ float s[64], ss[64];
  const int b = blockIdx.x;
  float2 p = partials[b * LN_BLOCKS + threadIdx.x];
  s[threadIdx.x] = p.x; ss[threadIdx.x] = p.y; __syncthreads();
  for (int st = 32; st > 0; st >>= 1) {
    if ((int)threadIdx.x < st) { s[threadIdx.x] += s[threadIdx.x + st];
                                 ss[threadIdx.x] += ss[threadIdx.x + st]; }
    __syncthreads();
  }
  if (threadIdx.x == 0) {
    const float inv = 1.0f / ((float)CDIM * (float)NTOK);
    float mu = s[0] * inv;
    float var = ss[0] * inv - mu * mu;
    float2 o; o.x = mu; o.y = rsqrtf(var + 1e-5f);
    stats[b] = o;
  }
}

__global__ __launch_bounds__(256)
void ln_normalize(float* __restrict__ xio, const float* __restrict__ w,
                  const float* __restrict__ bias, const float2* __restrict__ stats) {
  size_t i4 = (size_t)blockIdx.x * 256 + threadIdx.x;   // float4 index
  const size_t per4 = (size_t)CDIM * NTOK / 4;
  size_t b = i4 / per4, cn4 = i4 % per4;
  float2 st = stats[b];
  float4 v  = ((const float4*)xio)[i4];
  float4 wv = ((const float4*)w)[cn4];
  float4 bv = ((const float4*)bias)[cn4];
  v.x = (v.x - st.x) * st.y * wv.x + bv.x;
  v.y = (v.y - st.x) * st.y * wv.y + bv.y;
  v.z = (v.z - st.x) * st.y * wv.z + bv.z;
  v.w = (v.w - st.x) * st.y * wv.w + bv.w;
  ((float4*)xio)[i4] = v;
}

// ---------------------------------------------------------------------------
extern "C" void kernel_launch(void* const* d_in, const int* in_sizes, int n_in,
                              void* d_out, int out_size, void* d_ws, size_t ws_size,
                              hipStream_t stream) {
  (void)in_sizes; (void)n_in; (void)out_size; (void)ws_size;
  const float* x      = (const float*)d_in[0];
  const float* qkv_w  = (const float*)d_in[1];
  const float* proj_w = (const float*)d_in[2];
  const float* proj_b = (const float*)d_in[3];
  const float* ln1_w  = (const float*)d_in[4];
  const float* ln1_b  = (const float*)d_in[5];
  const float* ln2_w  = (const float*)d_in[6];
  const float* ln2_b  = (const float*)d_in[7];
  const float* mlp_w1 = (const float*)d_in[8];
  const float* mlp_b1 = (const float*)d_in[9];
  const float* mlp_w2 = (const float*)d_in[10];
  const float* mlp_b2 = (const float*)d_in[11];
  float* out = (float*)d_out;

  char* ws = (char*)d_ws;
  size_t off = 0;
  auto alloc = [&](size_t bytes) { size_t p = off; off = (off + bytes + 255) & ~(size_t)255; return p; };

  const size_t bigBytes = (size_t)BDIM * 384 * NTOK * 4;   // 384 MB: qkv -> x1|hidT
  const size_t tmBytes  = (size_t)BDIM * NTOK * CDIM * 2;  // 64 MB token-major bf16
  size_t bigOff  = alloc(bigBytes);
  size_t tmOff   = alloc(tmBytes);
  size_t wqOff   = alloc((size_t)384 * CDIM * 2);
  size_t w1Off   = alloc((size_t)MLPD * CDIM * 2);
  size_t w2Off   = alloc((size_t)CDIM * MLPD * 2);
  size_t MbOff   = alloc((size_t)BDIM * CDIM * CDIM * 2);
  size_t attnOff = alloc((size_t)BDIM * 8 * 256 * 4);
  size_t partOff = alloc((size_t)BDIM * LN_BLOCKS * 8);
  size_t statOff = alloc((size_t)BDIM * 8);

  float*          qkvBuf = (float*)(ws + bigOff);
  float*          x1     = qkvBuf;                          // 128 MB reuse
  unsigned short* hidT   = (unsigned short*)(ws + bigOff + (size_t)BDIM * CDIM * NTOK * 4);
  unsigned short* tm     = (unsigned short*)(ws + tmOff);
  unsigned short* wq     = (unsigned short*)(ws + wqOff);
  unsigned short* w1     = (unsigned short*)(ws + w1Off);
  unsigned short* w2     = (unsigned short*)(ws + w2Off);
  unsigned short* Mb     = (unsigned short*)(ws + MbOff);
  float*          attn   = (float*)(ws + attnOff);
  float2*         part   = (float2*)(ws + partOff);
  float2*         stats  = (float2*)(ws + statOff);

  const dim3 blk256(256);
  const dim3 trGrid(NTOK / 32, CDIM / 32, BDIM);

  // 1. weights -> bf16
  cvt_f32_bf16_kernel<<<(384 * CDIM) / 256, blk256, 0, stream>>>(qkv_w, wq, 384 * CDIM);
  cvt_f32_bf16_kernel<<<(MLPD * CDIM) / 256, blk256, 0, stream>>>(mlp_w1, w1, MLPD * CDIM);
  cvt_f32_bf16_kernel<<<(CDIM * MLPD) / 256, blk256, 0, stream>>>(mlp_w2, w2, CDIM * MLPD);

  // 2. xT (token-major bf16)
  transpose_cm_to_tm_bf16<<<trGrid, blk256, 0, stream>>>(x, (long long)CDIM * NTOK, CDIM, tm);

  // 3. GEMM1: qkv = qkv_w @ x   -> channel-major f32
  gemm_bf16_wmma<CDIM><<<dim3(NTOK / 256, 384 / 64, BDIM), blk256, 0, stream>>>(
      wq, 0, tm, 384, nullptr, nullptr, qkvBuf, nullptr, 0);

  // 4. channel-channel attention scores + softmax (fp32 WMMA, K=32768)
  attn_scores_softmax<<<BDIM * 8, blk256, 0, stream>>>(qkvBuf, attn);

  // 5. vT (token-major bf16) from qkv v-slice (overwrites dead xT)
  transpose_cm_to_tm_bf16<<<trGrid, blk256, 0, stream>>>(
      qkvBuf + (size_t)256 * NTOK, (long long)384 * NTOK, CDIM, tm);

  // 6. fuse proj_w with block-diagonal attention: M_b (bf16, per-batch)
  fuse_proj_attn<<<BDIM, blk256, 0, stream>>>(proj_w, attn, Mb);

  // 7. GEMM2: x1_pre = M_b @ v + proj_b + x   (writes over qkv region)
  gemm_bf16_wmma<CDIM><<<dim3(NTOK / 256, CDIM / 64, BDIM), blk256, 0, stream>>>(
      Mb, (long long)CDIM * CDIM, tm, CDIM, proj_b, x, x1, nullptr, 0);

  // 8. LayerNorm1 (full-tensor, in-place)
  ln_reduce<<<dim3(LN_BLOCKS, BDIM), blk256, 0, stream>>>(x1, part);
  ln_finalize<<<BDIM, dim3(64), 0, stream>>>(part, stats);
  ln_normalize<<<(BDIM * CDIM * NTOK / 4) / 256, blk256, 0, stream>>>(x1, ln1_w, ln1_b, stats);

  // 9. x1T (token-major bf16, overwrites dead vT)
  transpose_cm_to_tm_bf16<<<trGrid, blk256, 0, stream>>>(x1, (long long)CDIM * NTOK, CDIM, tm);

  // 10. GEMM3: hidT = gelu(mlp_w1 @ x1 + b1)  -> token-major bf16 via LDS epilogue
  gemm_bf16_wmma<CDIM><<<dim3(NTOK / 256, MLPD / 64, BDIM), blk256, 0, stream>>>(
      w1, 0, tm, MLPD, mlp_b1, nullptr, nullptr, hidT, 1);

  // 11. GEMM4: out = mlp_w2 @ hid + b2 + x1  -> d_out (channel-major f32)
  gemm_bf16_wmma<MLPD><<<dim3(NTOK / 256, CDIM / 64, BDIM), blk256, 0, stream>>>(
      w2, 0, hidT, CDIM, mlp_b2, x1, out, nullptr, 0);

  // 12. LayerNorm2 (in-place on d_out)
  ln_reduce<<<dim3(LN_BLOCKS, BDIM), blk256, 0, stream>>>(out, part);
  ln_finalize<<<BDIM, dim3(64), 0, stream>>>(part, stats);
  ln_normalize<<<(BDIM * CDIM * NTOK / 4) / 256, blk256, 0, stream>>>(out, ln2_w, ln2_b, stats);
}
